// FreqGuidedAttn_28114855919738
// MI455X (gfx1250) — compile-verified, matrix operand order checked by reading, not run on hardware
//
#include <hip/hip_runtime.h>
#include <hip/hip_bf16.h>
#include <math.h>

// FreqGuidedAttn for MI455X (gfx1250).
// 2-D DFT-as-matmul on the bf16 XDL pipe: V_WMMA_F32_16X16X32_BF16 with
// error-free hi/lo f32->bf16 split (3 WMMAs per 32-deep K chunk, f32 accum).
// x_hp = x - x_lp algebraically, so only one inverse-transform pipeline.

#define NPX 128           // H == W == 128
#define LSTR 136          // padded LDS row stride in bf16 elems (272B: 16B-aligned, conflict-free)
#define BATCH 16
#define NCH 128
#define MAX_FREQ_F 0.70710678118654752f   // 0.5*sqrt(2)
#define LO_OFF (128 * LSTR)               // offset of the "lo" plane inside a logical buffer

typedef __bf16 bf16_t;
typedef __attribute__((ext_vector_type(16))) __bf16 v16bf;
typedef __attribute__((ext_vector_type(8)))  float  v8f;

union FragU { uint4 q[2]; v16bf v; };

// -------- workspace layout --------
// bytes [0 .. 131072): Chi/Clo/Shi/Slo bf16 DFT matrices (4 x 16384 x 2B)
// floats from index 32768: x_lp, stats, attention maps.  Needs ~141 MB total.
#define OFF_XLP  ((size_t)32768)
#define N_XLP    ((size_t)BATCH * NCH * NPX * NPX)          // 33,554,432
#define OFF_LPST (OFF_XLP + N_XLP)
#define OFF_HPST (OFF_LPST + (size_t)BATCH * 2 * NPX * NPX)
#define OFF_ALP  (OFF_HPST + (size_t)BATCH * 2 * NPX * NPX)
#define OFF_AHP  (OFF_ALP + (size_t)BATCH * NPX * NPX)

// ---------------- bf16 split helpers ----------------
__device__ __forceinline__ unsigned short bf16_rne(float v) {
  unsigned u = __float_as_uint(v);
  u += 0x7fffu + ((u >> 16) & 1u);
  return (unsigned short)(u >> 16);
}
__device__ __forceinline__ void split_bf16(float v, unsigned short& h, unsigned short& l) {
  h = bf16_rne(v);
  float hv = __uint_as_float((unsigned)h << 16);
  l = bf16_rne(v - hv);
}

// ---------------- WMMA helpers ----------------
__device__ __forceinline__ v8f wmma_bf(v16bf a, v16bf b, v8f c) {
  return __builtin_amdgcn_wmma_f32_16x16x32_bf16(false, a, false, b, (short)0, c, false, false);
}
// acc += (Ahi+Alo)*(Bhi+Blo) ~= Ahi*Bhi + Ahi*Blo + Alo*Bhi   (f32 accumulate)
__device__ __forceinline__ v8f mul3(v16bf ah, v16bf al, v16bf bh, v16bf bl, v8f acc) {
  acc = wmma_bf(ah, bh, acc);
  acc = wmma_bf(ah, bl, acc);
  acc = wmma_bf(al, bh, acc);
  return acc;
}
__device__ __forceinline__ v16bf fneg(v16bf a) {
  FragU f; f.v = a;
#pragma unroll
  for (int i = 0; i < 2; ++i) {
    f.q[i].x ^= 0x80008000u; f.q[i].y ^= 0x80008000u;
    f.q[i].z ^= 0x80008000u; f.q[i].w ^= 0x80008000u;
  }
  return f.v;
}

// 16-bit A/B fragment: lane half kh selects K {koff..koff+7} and {koff+16..koff+23},
// each 16B contiguous -> two aligned b128 loads.
__device__ __forceinline__ v16bf frag_lds(const unsigned short* base, int row, int koff) {
  FragU f;
  f.q[0] = *(const uint4*)(base + row * LSTR + koff);
  f.q[1] = *(const uint4*)(base + row * LSTR + koff + 16);
  return f.v;
}
__device__ __forceinline__ v16bf frag_glb(const unsigned short* __restrict__ base, int row, int koff) {
  FragU f;
  f.q[0] = *(const uint4*)(base + row * NPX + koff);
  f.q[1] = *(const uint4*)(base + row * NPX + koff + 16);
  return f.v;
}

__device__ __forceinline__ void pack8_store(unsigned short* p, const unsigned short* s) {
  uint4 q;
  q.x = (unsigned)s[0] | ((unsigned)s[1] << 16);
  q.y = (unsigned)s[2] | ((unsigned)s[3] << 16);
  q.z = (unsigned)s[4] | ((unsigned)s[5] << 16);
  q.w = (unsigned)s[6] | ((unsigned)s[7] << 16);
  *(uint4*)p = q;
}

// C/D layout: VGPR j = (row = m16 + 8*(lane>=16) + j, col = n16 + lane%16).
// Row-major hi/lo store (feeds later A-operands): scattered b16 writes.
__device__ __forceinline__ void store_rm(unsigned short* buf, v8f acc, int m, int n, int lane, float scale) {
  int row0 = m * 16 + (lane >> 4) * 8;
  int col  = n * 16 + (lane & 15);
#pragma unroll
  for (int j = 0; j < 8; ++j) {
    unsigned short h, l;
    split_bf16(acc[j] * scale, h, l);
    buf[(row0 + j) * LSTR + col]          = h;
    buf[LO_OFF + (row0 + j) * LSTR + col] = l;
  }
}
// Transposed hi/lo store (feeds later B-operands): rows j contiguous -> one b128 per plane.
__device__ __forceinline__ void store_tr(unsigned short* buf, v8f acc, const float* hmul, int m, int n, int lane) {
  int row0 = m * 16 + (lane >> 4) * 8;
  int col  = n * 16 + (lane & 15);
  unsigned short hs[8], ls[8];
#pragma unroll
  for (int j = 0; j < 8; ++j) split_bf16(acc[j] * hmul[j], hs[j], ls[j]);
  pack8_store(buf + col * LSTR + row0, hs);
  pack8_store(buf + LO_OFF + col * LSTR + row0, ls);
}

// Butterworth LP response at frequency bin (kr,kc): H = D0^4/(D0^4 + D^4), order 2.
__device__ __forceinline__ float butter_lp(int kr, int kc, float d0p4) {
  float fu = (float)(kr <= 64 ? kr : NPX - kr) * (1.0f / NPX);
  float fv = (float)(kc <= 64 ? kc : NPX - kc) * (1.0f / NPX);
  float d2 = fmaf(fu, fu, fv * fv);
  return d0p4 / fmaf(d2, d2, d0p4);
}

// ---------------- kernel 0: DFT constants (bf16 hi/lo, symmetric) ----------------
__global__ void __launch_bounds__(256)
fga_init_consts(unsigned short* __restrict__ Chi, unsigned short* __restrict__ Clo,
                unsigned short* __restrict__ Shi, unsigned short* __restrict__ Slo) {
  int i = blockIdx.x * 256 + threadIdx.x;   // 16384 entries
  int k = i >> 7, n = i & 127;
  float t = 6.283185307179586f * (float)((k * n) & 127) * (1.0f / NPX);
  float s, c;
  __sincosf(t, &s, &c);
  unsigned short h, l;
  split_bf16(c, h, l); Chi[i] = h; Clo[i] = l;   // F = C - i S (symmetric)
  split_bf16(s, h, l); Shi[i] = h; Slo[i] = l;
}

// ---------------- kernel 1: per-image spectral filter via bf16 WMMA ----------------
// One workgroup (8 wave32) per (b,c) image; wave w owns output row-block m = w.
// LDS: 4 logical buffers x (hi+lo) x 128x136 bf16 = 272 KB of the 320 KB WGP LDS.
__global__ void __launch_bounds__(256)
fga_filter_kernel(const float* __restrict__ x, const float* __restrict__ alpha,
                  const unsigned short* __restrict__ Chi, const unsigned short* __restrict__ Clo,
                  const unsigned short* __restrict__ Shi, const unsigned short* __restrict__ Slo,
                  float* __restrict__ xlp) {
  extern __shared__ unsigned short sh[];
  unsigned short* L0 = sh;                    // x^T   -> Yr^T
  unsigned short* L1 = sh + 2 * 128 * LSTR;   // Ar    -> Zr
  unsigned short* L2 = sh + 4 * 128 * LSTR;   // Ai    -> Zi
  unsigned short* L3 = sh + 6 * 128 * LSTR;   //          Yi^T

  const int img  = blockIdx.x;                // b*NCH + c
  const int cch  = img & (NCH - 1);
  const int tid  = threadIdx.x;
  const int m    = tid >> 5;                  // wave id == output row-block
  const int lane = tid & 31;
  const int l16  = lane & 15;
  const int kh   = lane >> 4;

  const float* xg = x + (size_t)img * (NPX * NPX);
  float* og = xlp + (size_t)img * (NPX * NPX);

  float d0   = MAX_FREQ_F / (1.0f + __expf(-alpha[cch]));
  float d0sq = d0 * d0;
  float d0p4 = d0sq * d0sq;

  // Stage x into LDS transposed + hi/lo split: thread = (col, h-half).
  {
    int col = tid & 127;
    int h0  = (tid >> 7) * 64;
    for (int hb = 0; hb < 8; ++hb) {
      int hbase = h0 + hb * 8;
      unsigned short hs[8], ls[8];
#pragma unroll
      for (int j = 0; j < 8; ++j)
        split_bf16(xg[(hbase + j) * NPX + col], hs[j], ls[j]);
      pack8_store(L0 + col * LSTR + hbase, hs);
      pack8_store(L0 + LO_OFF + col * LSTR + hbase, ls);
    }
  }
  __syncthreads();

  const int arow = m * 16 + l16;

  // ---- Stage 1: A = F*x -> Ar = C*x (L1), Ai = -(S*x) (L2); A-op = consts, B-op = x^T.
  for (int g = 0; g < 2; ++g) {
    v8f ar[4] = {}, ai[4] = {};
    for (int kc = 0; kc < NPX; kc += 32) {
      int koff = kc + 8 * kh;
      v16bf ch = frag_glb(Chi, arow, koff), cl = frag_glb(Clo, arow, koff);
      v16bf sH = frag_glb(Shi, arow, koff), sL = frag_glb(Slo, arow, koff);
#pragma unroll
      for (int t = 0; t < 4; ++t) {
        int brow = (g * 4 + t) * 16 + l16;
        v16bf xh = frag_lds(L0, brow, koff), xl = frag_lds(L0 + LO_OFF, brow, koff);
        ar[t] = mul3(ch, cl, xh, xl, ar[t]);
        ai[t] = mul3(sH, sL, xh, xl, ai[t]);
      }
    }
#pragma unroll
    for (int t = 0; t < 4; ++t) {
      store_rm(L1, ar[t], m, g * 4 + t, lane, 1.0f);
      store_rm(L2, ai[t], m, g * 4 + t, lane, -1.0f);
    }
  }
  __syncthreads();

  // ---- Stage 2: B = A*F^T -> Br = Ar*C + Ai*S, Bi = Ai*C - Ar*S.
  //      Butterworth filter applied in f32 at store; outputs stored transposed (Y^T).
  for (int g = 0; g < 2; ++g) {
    v8f br[4] = {}, bi[4] = {};
    for (int kc = 0; kc < NPX; kc += 32) {
      int koff = kc + 8 * kh;
      v16bf arh = frag_lds(L1, arow, koff), arl = frag_lds(L1 + LO_OFF, arow, koff);
      v16bf aih = frag_lds(L2, arow, koff), ail = frag_lds(L2 + LO_OFF, arow, koff);
      v16bf nrh = fneg(arh), nrl = fneg(arl);
#pragma unroll
      for (int t = 0; t < 4; ++t) {
        int brow = (g * 4 + t) * 16 + l16;
        v16bf bch = frag_glb(Chi, brow, koff), bcl = frag_glb(Clo, brow, koff);
        v16bf bsh = frag_glb(Shi, brow, koff), bsl = frag_glb(Slo, brow, koff);
        br[t] = mul3(arh, arl, bch, bcl, br[t]);
        br[t] = mul3(aih, ail, bsh, bsl, br[t]);
        bi[t] = mul3(aih, ail, bch, bcl, bi[t]);
        bi[t] = mul3(nrh, nrl, bsh, bsl, bi[t]);
      }
    }
#pragma unroll
    for (int t = 0; t < 4; ++t) {
      int n = g * 4 + t;
      int row0 = m * 16 + (lane >> 4) * 8;
      int col  = n * 16 + l16;
      float hm[8];
#pragma unroll
      for (int j = 0; j < 8; ++j) hm[j] = butter_lp(row0 + j, col, d0p4);
      store_tr(L0, br[t], hm, m, n, lane);   // Yr^T
      store_tr(L3, bi[t], hm, m, n, lane);   // Yi^T
    }
  }
  __syncthreads();

  // ---- Stage 3: Z = F^{-1}*Y -> Zr = C*Yr - S*Yi (L1), Zi = C*Yi + S*Yr (L2).
  for (int g = 0; g < 2; ++g) {
    v8f zr[4] = {}, zi[4] = {};
    for (int kc = 0; kc < NPX; kc += 32) {
      int koff = kc + 8 * kh;
      v16bf ch = frag_glb(Chi, arow, koff), cl = frag_glb(Clo, arow, koff);
      v16bf sH = frag_glb(Shi, arow, koff), sL = frag_glb(Slo, arow, koff);
      v16bf nsh = fneg(sH), nsl = fneg(sL);
#pragma unroll
      for (int t = 0; t < 4; ++t) {
        int brow = (g * 4 + t) * 16 + l16;
        v16bf yrh = frag_lds(L0, brow, koff), yrl = frag_lds(L0 + LO_OFF, brow, koff);
        v16bf yih = frag_lds(L3, brow, koff), yil = frag_lds(L3 + LO_OFF, brow, koff);
        zr[t] = mul3(ch, cl, yrh, yrl, zr[t]);
        zr[t] = mul3(nsh, nsl, yih, yil, zr[t]);
        zi[t] = mul3(ch, cl, yih, yil, zi[t]);
        zi[t] = mul3(sH, sL, yrh, yrl, zi[t]);
      }
    }
#pragma unroll
    for (int t = 0; t < 4; ++t) {
      store_rm(L1, zr[t], m, g * 4 + t, lane, 1.0f);
      store_rm(L2, zi[t], m, g * 4 + t, lane, 1.0f);
    }
  }
  __syncthreads();

  // ---- Stage 4: x_lp = (Zr*C - Zi*S)/16384 -> global.
  for (int g = 0; g < 2; ++g) {
    v8f o[4] = {};
    for (int kc = 0; kc < NPX; kc += 32) {
      int koff = kc + 8 * kh;
      v16bf zrh = frag_lds(L1, arow, koff), zrl = frag_lds(L1 + LO_OFF, arow, koff);
      v16bf zih = frag_lds(L2, arow, koff), zil = frag_lds(L2 + LO_OFF, arow, koff);
      v16bf nzh = fneg(zih), nzl = fneg(zil);
#pragma unroll
      for (int t = 0; t < 4; ++t) {
        int brow = (g * 4 + t) * 16 + l16;
        v16bf bch = frag_glb(Chi, brow, koff), bcl = frag_glb(Clo, brow, koff);
        v16bf bsh = frag_glb(Shi, brow, koff), bsl = frag_glb(Slo, brow, koff);
        o[t] = mul3(zrh, zrl, bch, bcl, o[t]);
        o[t] = mul3(nzh, nzl, bsh, bsl, o[t]);
      }
    }
#pragma unroll
    for (int t = 0; t < 4; ++t) {
      int row0 = m * 16 + (lane >> 4) * 8;
      int colo = (g * 4 + t) * 16 + l16;
#pragma unroll
      for (int j = 0; j < 8; ++j)
        og[(row0 + j) * NPX + colo] = o[t][j] * (1.0f / 16384.0f);
    }
  }
}

// ---------------- kernel 2: channel mean/max stats ----------------
__global__ void __launch_bounds__(256)
fga_stats_kernel(const float* __restrict__ x, const float* __restrict__ xlp,
                 float* __restrict__ lpst, float* __restrict__ hpst) {
  int pix = blockIdx.x * 256 + threadIdx.x;   // b*16384 + p
  int b = pix >> 14, p = pix & 16383;
  const float* xb = x + (size_t)b * NCH * 16384 + p;
  const float* lb = xlp + (size_t)b * NCH * 16384 + p;
  float slp = 0.f, shp = 0.f, mlp = -INFINITY, mhp = -INFINITY;
  for (int c = 0; c < NCH; ++c) {
    float xv = xb[(size_t)c * 16384];
    float lv = lb[(size_t)c * 16384];
    float hv = xv - lv;
    slp += lv; shp += hv;
    mlp = fmaxf(mlp, lv); mhp = fmaxf(mhp, hv);
  }
  lpst[(size_t)b * 2 * 16384 + p]         = slp * (1.0f / NCH);
  lpst[(size_t)b * 2 * 16384 + 16384 + p] = mlp;
  hpst[(size_t)b * 2 * 16384 + p]         = shp * (1.0f / NCH);
  hpst[(size_t)b * 2 * 16384 + 16384 + p] = mhp;
}

// ---------------- kernel 3: 7x7 conv (2->1 ch) + sigmoid, both branches ----------------
__global__ void __launch_bounds__(256)
fga_conv_kernel(const float* __restrict__ lpst, const float* __restrict__ hpst,
                const float* __restrict__ wlp, const float* __restrict__ blp,
                const float* __restrict__ whp, const float* __restrict__ bhp,
                float* __restrict__ alp, float* __restrict__ ahp) {
  __shared__ float wl[98], wh[98];
  if (threadIdx.x < 98) { wl[threadIdx.x] = wlp[threadIdx.x]; wh[threadIdx.x] = whp[threadIdx.x]; }
  __syncthreads();
  int pix = blockIdx.x * 256 + threadIdx.x;
  int b = pix >> 14, p = pix & 16383;
  int h = p >> 7, w = p & 127;
  float accl = 0.f, acch = 0.f;
  for (int s = 0; s < 2; ++s) {
    const float* lsrc = lpst + ((size_t)b * 2 + s) * 16384;
    const float* hsrc = hpst + ((size_t)b * 2 + s) * 16384;
    for (int u = 0; u < 7; ++u) {
      int hh = h + u - 3;
      if (hh < 0 || hh >= 128) continue;
      for (int v = 0; v < 7; ++v) {
        int ww = w + v - 3;
        if (ww < 0 || ww >= 128) continue;
        int wi = s * 49 + u * 7 + v;
        int si = hh * 128 + ww;
        accl = fmaf(lsrc[si], wl[wi], accl);
        acch = fmaf(hsrc[si], wh[wi], acch);
      }
    }
  }
  alp[pix] = 1.0f / (1.0f + __expf(-(accl + blp[0])));
  ahp[pix] = 1.0f / (1.0f + __expf(-(acch + bhp[0])));
}

// ---------------- kernel 4: fused recombination ----------------
__global__ void __launch_bounds__(256)
fga_combine_kernel(const float* __restrict__ x, const float* __restrict__ xlp,
                   const float* __restrict__ alp, const float* __restrict__ ahp,
                   const float* __restrict__ glp, const float* __restrict__ ghp,
                   float* __restrict__ out) {
  size_t e = (size_t)blockIdx.x * 256 + threadIdx.x;
  size_t i = e * 4;
  int p = (int)(i & 16383);
  int c = (int)((i >> 14) & 127);
  int b = (int)(i >> 21);
  float gl = glp[c], gh = ghp[c];
  const float4 xv = *(const float4*)(x + i);
  const float4 lv = *(const float4*)(xlp + i);
  const float4 al = *(const float4*)(alp + (size_t)b * 16384 + p);
  const float4 ah = *(const float4*)(ahp + (size_t)b * 16384 + p);
  float4 o;
  o.x = xv.x + (xv.x - lv.x) * ah.x * gh + lv.x * al.x * gl;
  o.y = xv.y + (xv.y - lv.y) * ah.y * gh + lv.y * al.y * gl;
  o.z = xv.z + (xv.z - lv.z) * ah.z * gh + lv.z * al.z * gl;
  o.w = xv.w + (xv.w - lv.w) * ah.w * gh + lv.w * al.w * gl;
  *(float4*)(out + i) = o;
}

extern "C" void kernel_launch(void* const* d_in, const int* in_sizes, int n_in,
                              void* d_out, int out_size, void* d_ws, size_t ws_size,
                              hipStream_t stream) {
  const float* x     = (const float*)d_in[0];
  const float* alpha = (const float*)d_in[1];
  const float* wlp   = (const float*)d_in[2];
  const float* blp   = (const float*)d_in[3];
  const float* whp   = (const float*)d_in[4];
  const float* bhp   = (const float*)d_in[5];
  const float* glp   = (const float*)d_in[6];
  const float* ghp   = (const float*)d_in[7];
  float* out = (float*)d_out;

  unsigned short* Chi = (unsigned short*)d_ws;
  unsigned short* Clo = Chi + 16384;
  unsigned short* Shi = Clo + 16384;
  unsigned short* Slo = Shi + 16384;
  float* ws   = (float*)d_ws;
  float* xlp  = ws + OFF_XLP;
  float* lpst = ws + OFF_LPST;
  float* hpst = ws + OFF_HPST;
  float* alp  = ws + OFF_ALP;
  float* ahp  = ws + OFF_AHP;

  fga_init_consts<<<64, 256, 0, stream>>>(Chi, Clo, Shi, Slo);

  size_t lds_bytes = (size_t)8 * 128 * LSTR * sizeof(unsigned short);   // 278,528 B of 320 KB
  fga_filter_kernel<<<BATCH * NCH, 256, lds_bytes, stream>>>(x, alpha, Chi, Clo, Shi, Slo, xlp);

  fga_stats_kernel<<<(BATCH * NPX * NPX) / 256, 256, 0, stream>>>(x, xlp, lpst, hpst);

  fga_conv_kernel<<<(BATCH * NPX * NPX) / 256, 256, 0, stream>>>(lpst, hpst, wlp, blp, whp, bhp, alp, ahp);

  fga_combine_kernel<<<(int)(N_XLP / 4 / 256), 256, 0, stream>>>(x, xlp, alp, ahp, glp, ghp, out);
}
